// Multi_Head_Attention_68178310856819
// MI455X (gfx1250) — matrix operand order repaired and checked by reading, not compiled, not run on hardware
//
#include <hip/hip_runtime.h>
#include <hip/hip_bf16.h>

#define D_MODEL 1024
#define HEADS   16
#define DKH     64
#define BATCH   2
#define SEQ     2048
#define MROWS   (BATCH * SEQ)   // 4096

typedef _Float16 h8  __attribute__((ext_vector_type(8)));
typedef _Float16 h16 __attribute__((ext_vector_type(16)));
typedef float    f8  __attribute__((ext_vector_type(8)));
typedef float    f4v __attribute__((ext_vector_type(4)));

__device__ __forceinline__ f8 wmma_f16(h16 a, h16 b, f8 c) {
    // D = A(16x32 f16) * B(32x16 f16) + C(16x16 f32)
    return __builtin_amdgcn_wmma_f32_16x16x32_f16(
        false, a, false, b, (short)0, c, false, false);
}

// A-fragment (16x32 f16) from a contiguous f16 row.
// lane m = lane&15 ; koff = (lane>>4)*8 ; elements [0..15]=row[2*koff..]
// (contiguous k = 16*half + s, matching the B fills).
__device__ __forceinline__ h16 load_a_f16(const _Float16* row, int koff) {
    h8 lo = *(const h8*)(row + 2 * koff);
    h8 hi = *(const h8*)(row + 2 * koff + 8);
    return __builtin_shufflevector(lo, hi, 0,1,2,3,4,5,6,7,8,9,10,11,12,13,14,15);
}

// ---------------------------------------------------------------------------
// Kernel 0: elementwise fp32 -> f16 (8 elements / thread)
// ---------------------------------------------------------------------------
__global__ __launch_bounds__(256) void cvt_f16_kernel(
    const float* __restrict__ src, _Float16* __restrict__ dst)
{
    const size_t i = ((size_t)blockIdx.x * 256 + threadIdx.x) * 8;
    f4v a = *(const f4v*)(src + i);
    f4v b = *(const f4v*)(src + i + 4);
    h8 o;
#pragma unroll
    for (int j = 0; j < 4; ++j) {
        o[j]     = (_Float16)a[j];
        o[4 + j] = (_Float16)b[j];
    }
    *(h8*)(dst + i) = o;
}

// ---------------------------------------------------------------------------
// Kernel 1: q/k/v = X @ W^T  (all-f16 inputs, attention-friendly layouts)
// Each wave computes a 32x64 tile.  block = 128 (4 waves) -> 128 rows.
// grid.x = (MROWS/128)*16, grid.y = 3 (Q,K,V)
// ---------------------------------------------------------------------------
__global__ __launch_bounds__(128) void qkv_proj_kernel(
    const _Float16* __restrict__ xq, const _Float16* __restrict__ xk,
    const _Float16* __restrict__ xv,
    const _Float16* __restrict__ wq, const _Float16* __restrict__ wk,
    const _Float16* __restrict__ wv,
    _Float16* __restrict__ qh, _Float16* __restrict__ kh, _Float16* __restrict__ vt)
{
    const int sel = blockIdx.y;
    const _Float16* X = sel == 0 ? xq : (sel == 1 ? xk : xv);
    const _Float16* W = sel == 0 ? wq : (sel == 1 ? wk : wv);

    const int nblk = blockIdx.x & 15;   // 16 column blocks of 64
    const int mblk = blockIdx.x >> 4;   // 32 row blocks of 128
    const int wave = threadIdx.x >> 5;
    const int lane = threadIdx.x & 31;
    const int lo16 = lane & 15;
    const int hl   = lane >> 4;
    const int koff = hl * 8;

    const int m0 = mblk * 128 + wave * 32;
    const int n0 = nblk * 64;

    const _Float16* xrow0 = X + (size_t)(m0 + lo16) * D_MODEL;
    const _Float16* xrow1 = X + (size_t)(m0 + 16 + lo16) * D_MODEL;

    f8 acc[2][4] = {};
#pragma unroll 2
    for (int kt = 0; kt < D_MODEL / 32; ++kt) {
        const int kb = kt * 32;
        h16 a0 = load_a_f16(xrow0 + kb, koff);
        h16 a1 = load_a_f16(xrow1 + kb, koff);
#pragma unroll
        for (int nt = 0; nt < 4; ++nt) {
            const _Float16* wrow =
                W + (size_t)(n0 + nt * 16 + lo16) * D_MODEL + kb + hl * 16;
            h16 bf = *(const h16*)wrow;
            acc[0][nt] = wmma_f16(a0, bf, acc[0][nt]);
            acc[1][nt] = wmma_f16(a1, bf, acc[1][nt]);
        }
    }

#pragma unroll
    for (int t = 0; t < 2; ++t) {
#pragma unroll
        for (int nt = 0; nt < 4; ++nt) {
            const int n  = n0 + nt * 16 + lo16;
            const int h  = n >> 6;
            const int dk = n & 63;
#pragma unroll
            for (int r = 0; r < 8; ++r) {
                const int row = m0 + t * 16 + r + hl * 8;  // global row
                const int b   = row >> 11;                 // / SEQ
                const int s   = row & (SEQ - 1);
                const _Float16 val = (_Float16)acc[t][nt][r];
                if (sel == 2) {
                    vt[((size_t)(b * HEADS + h) * DKH + dk) * SEQ + s] = val;
                } else {
                    _Float16* o = sel ? kh : qh;
                    o[((size_t)(b * HEADS + h) * SEQ + s) * DKH + dk] = val;
                }
            }
        }
    }
}

// ---------------------------------------------------------------------------
// Kernel 2: flash attention.  block = 128 (4 waves), each wave = 16 q-rows.
// grid = B*H*(SEQ/64)
// ---------------------------------------------------------------------------
__global__ __launch_bounds__(128) void attn_kernel(
    const _Float16* __restrict__ qh, const _Float16* __restrict__ kh,
    const _Float16* __restrict__ vt, const int* __restrict__ mask,
    _Float16* __restrict__ attn_out)
{
    const int qblocks = SEQ / 64;
    const int qblk = blockIdx.x % qblocks;
    const int bh   = blockIdx.x / qblocks;
    const int h    = bh % HEADS;
    const int b    = bh / HEADS;

    const int wave = threadIdx.x >> 5;
    const int lane = threadIdx.x & 31;
    const int lo16 = lane & 15;
    const int hl   = lane >> 4;
    const int koff = hl * 8;
    const int s0   = qblk * 64 + wave * 16;

    const _Float16* qbase = qh + (size_t)(b * HEADS + h) * SEQ * DKH;
    const _Float16* kbase = kh + (size_t)(b * HEADS + h) * SEQ * DKH;
    const _Float16* vbase = vt + (size_t)(b * HEADS + h) * DKH * SEQ;
    const int* mp = mask + b * SEQ;

    // Q A-fragments (contraction dk 0..31 and 32..63)
    const _Float16* qrow = qbase + (size_t)(s0 + lo16) * DKH;
    h16 aq0 = load_a_f16(qrow,      koff);
    h16 aq1 = load_a_f16(qrow + 32, koff);

    f8 acc[4] = {};
    float mrow[8], lsum[8];
#pragma unroll
    for (int r = 0; r < 8; ++r) { mrow[r] = -1e30f; lsum[r] = 0.0f; }

    // Per-wave private P tile, padded stride 40 halves (80 B: 16B-aligned,
    // conflict-free across the 16 row-banks).
    __shared__ __align__(16) _Float16 plds[4][16 * 40];
    _Float16* myp = plds[wave];

    const float scale = 0.125f;  // 1/sqrt(64)

    for (int kt = 0; kt < SEQ / 32; ++kt) {
        const int key0 = kt * 32;

        // ---- scores: two 16x16 tiles (keys key0..+15 and key0+16..+31) ----
        f8 sA = {}, sB = {};
        {
            const _Float16* krA = kbase + (size_t)(key0 + lo16) * DKH;
            const _Float16* krB = kbase + (size_t)(key0 + 16 + lo16) * DKH;
            h16 bA0 = *(const h16*)(krA + hl * 16);
            h16 bA1 = *(const h16*)(krA + 32 + hl * 16);
            sA = wmma_f16(aq0, bA0, sA);
            sA = wmma_f16(aq1, bA1, sA);
            h16 bB0 = *(const h16*)(krB + hl * 16);
            h16 bB1 = *(const h16*)(krB + 32 + hl * 16);
            sB = wmma_f16(aq0, bB0, sB);
            sB = wmma_f16(aq1, bB1, sB);
        }
        const int mA = mp[key0 + lo16];
        const int mB = mp[key0 + 16 + lo16];

        // ---- online softmax (row lives across 16 lanes of a half-wave) ----
#pragma unroll
        for (int r = 0; r < 8; ++r) {
            float xA = mA ? sA[r] * scale : -1e30f;
            float xB = mB ? sB[r] * scale : -1e30f;
            float t = fmaxf(xA, xB);
            t = fmaxf(t, __shfl_xor(t, 1, 16));
            t = fmaxf(t, __shfl_xor(t, 2, 16));
            t = fmaxf(t, __shfl_xor(t, 4, 16));
            t = fmaxf(t, __shfl_xor(t, 8, 16));
            const float newm = fmaxf(mrow[r], t);
            const float corr = __expf(mrow[r] - newm);
            const float pA = __expf(xA - newm);
            const float pB = __expf(xB - newm);
            float ps = pA + pB;
            ps += __shfl_xor(ps, 1, 16);
            ps += __shfl_xor(ps, 2, 16);
            ps += __shfl_xor(ps, 4, 16);
            ps += __shfl_xor(ps, 8, 16);
            lsum[r] = lsum[r] * corr + ps;
            mrow[r] = newm;
            acc[0][r] *= corr; acc[1][r] *= corr;
            acc[2][r] *= corr; acc[3][r] *= corr;
            // D-layout -> row-major P tile in per-wave LDS region (f16)
            const int m = r + hl * 8;
            myp[m * 40 + lo16]      = (_Float16)pA;
            myp[m * 40 + 16 + lo16] = (_Float16)pB;
        }

        // LDS region is wave-private and DS ops are in-order per wave:
        // a DScnt wait is enough (no block barrier needed).
        asm volatile("s_wait_dscnt 0x0" ::: "memory");

        // ---- re-read P in A-fragment layout, then P @ V ----
        h16 ap;
        {
            const _Float16* prow = myp + lo16 * 40;
            h8 lo = *(const h8*)(prow + 2 * koff);
            h8 hi = *(const h8*)(prow + 2 * koff + 8);
            ap = __builtin_shufflevector(lo, hi, 0,1,2,3,4,5,6,7,8,9,10,11,12,13,14,15);
        }
#pragma unroll
        for (int nt = 0; nt < 4; ++nt) {
            const _Float16* vr =
                vbase + (size_t)(nt * 16 + lo16) * SEQ + key0 + hl * 16;
            h16 bv = *(const h16*)vr;
            acc[nt] = wmma_f16(ap, bv, acc[nt]);
        }
        // In-order LDS per wave: next iteration's stores cannot pass the
        // loads above, so no extra wait is required here.
    }

    // ---- epilogue: divide by softmax denominator, store f16 [b,s,d] ----
#pragma unroll
    for (int nt = 0; nt < 4; ++nt) {
#pragma unroll
        for (int r = 0; r < 8; ++r) {
            const float o = acc[nt][r] / lsum[r];
            const int s = s0 + r + hl * 8;
            attn_out[(size_t)(b * SEQ + s) * D_MODEL + h * 64 + nt * 16 + lo16] =
                (_Float16)o;
        }
    }
}

// ---------------------------------------------------------------------------
// Kernel 3: proj = attn_out(f16) @ Wo^T(f16) -> fp32 scratch
// Each wave computes a 32x64 tile (same shape as kernel 1).
// ---------------------------------------------------------------------------
__global__ __launch_bounds__(128) void oproj_kernel(
    const _Float16* __restrict__ O, const _Float16* __restrict__ Wo,
    float* __restrict__ proj)
{
    const int nblk = blockIdx.x & 15;
    const int mblk = blockIdx.x >> 4;
    const int wave = threadIdx.x >> 5;
    const int lane = threadIdx.x & 31;
    const int lo16 = lane & 15;
    const int hl   = lane >> 4;
    const int koff = hl * 8;

    const int m0 = mblk * 128 + wave * 32;
    const int n0 = nblk * 64;

    const _Float16* orow0 = O + (size_t)(m0 + lo16) * D_MODEL;
    const _Float16* orow1 = O + (size_t)(m0 + 16 + lo16) * D_MODEL;

    f8 acc[2][4] = {};
#pragma unroll 2
    for (int kt = 0; kt < D_MODEL / 32; ++kt) {
        const int kb = kt * 32;
        h16 a0 = load_a_f16(orow0 + kb, koff);
        h16 a1 = load_a_f16(orow1 + kb, koff);
#pragma unroll
        for (int nt = 0; nt < 4; ++nt) {
            const _Float16* wrow =
                Wo + (size_t)(n0 + nt * 16 + lo16) * D_MODEL + kb + hl * 16;
            h16 bf = *(const h16*)wrow;
            acc[0][nt] = wmma_f16(a0, bf, acc[0][nt]);
            acc[1][nt] = wmma_f16(a1, bf, acc[1][nt]);
        }
    }

#pragma unroll
    for (int t = 0; t < 2; ++t) {
#pragma unroll
        for (int nt = 0; nt < 4; ++nt) {
            const int n = n0 + nt * 16 + lo16;
#pragma unroll
            for (int r = 0; r < 8; ++r) {
                const int row = m0 + t * 16 + r + hl * 8;
                proj[(size_t)row * D_MODEL + n] = acc[t][nt][r];
            }
        }
    }
}

// ---------------------------------------------------------------------------
// Kernel 4: out = LayerNorm(proj + Q)  (one row per block, 256 threads)
// ---------------------------------------------------------------------------
__global__ __launch_bounds__(256) void ln_kernel(
    const float* __restrict__ proj, const float* __restrict__ Qin,
    const float* __restrict__ gamma, const float* __restrict__ beta,
    float* __restrict__ out)
{
    const int row = blockIdx.x;
    const int tid = threadIdx.x;
    const float* p = proj + (size_t)row * D_MODEL;
    const float* q = Qin  + (size_t)row * D_MODEL;

    float x[4];
    float s = 0.0f, s2 = 0.0f;
#pragma unroll
    for (int i = 0; i < 4; ++i) {
        const int c = tid + i * 256;
        const float v = p[c] + q[c];
        x[i] = v; s += v; s2 += v * v;
    }

    __shared__ float red0[256];
    __shared__ float red1[256];
    red0[tid] = s; red1[tid] = s2;
    __syncthreads();
    for (int off = 128; off > 0; off >>= 1) {
        if (tid < off) { red0[tid] += red0[tid + off]; red1[tid] += red1[tid + off]; }
        __syncthreads();
    }
    const float mu   = red0[0] * (1.0f / D_MODEL);
    const float var  = red1[0] * (1.0f / D_MODEL) - mu * mu;
    const float rinv = rsqrtf(var + 1e-5f);

#pragma unroll
    for (int i = 0; i < 4; ++i) {
        const int c = tid + i * 256;
        out[(size_t)row * D_MODEL + c] = (x[i] - mu) * rinv * gamma[c] + beta[c];
    }
}

// ---------------------------------------------------------------------------
extern "C" void kernel_launch(void* const* d_in, const int* in_sizes, int n_in,
                              void* d_out, int out_size, void* d_ws, size_t ws_size,
                              hipStream_t stream)
{
    const float* Q    = (const float*)d_in[0];
    const float* K    = (const float*)d_in[1];
    const float* V    = (const float*)d_in[2];
    const int*   mask = (const int*)  d_in[3];
    const float* Wq   = (const float*)d_in[4];
    const float* Wk   = (const float*)d_in[5];
    const float* Wv   = (const float*)d_in[6];
    const float* Wo   = (const float*)d_in[7];
    const float* gam  = (const float*)d_in[8];
    const float* bet  = (const float*)d_in[9];
    float* out = (float*)d_out;

    const size_t X_SZ = (size_t)MROWS * D_MODEL * sizeof(_Float16);   // 8 MB
    const size_t W_SZ = (size_t)D_MODEL * D_MODEL * sizeof(_Float16); // 2 MB
    char* ws = (char*)d_ws;
    _Float16* xq = (_Float16*)(ws);                 // [0,  8 MB)
    _Float16* xk = (_Float16*)(ws + 1 * X_SZ);      // [8, 16 MB)
    _Float16* xv = (_Float16*)(ws + 2 * X_SZ);      // [16,24 MB)
    _Float16* wq = (_Float16*)(ws + 3 * X_SZ);      // [24,26 MB)
    _Float16* wk = (_Float16*)(ws + 3 * X_SZ + 1 * W_SZ);
    _Float16* wv = (_Float16*)(ws + 3 * X_SZ + 2 * W_SZ);
    _Float16* wo = (_Float16*)(ws + 3 * X_SZ + 3 * W_SZ);
    _Float16* qh = (_Float16*)(ws + 4 * X_SZ);      // [32,40 MB)
    _Float16* kh = (_Float16*)(ws + 5 * X_SZ);      // [40,48 MB)
    _Float16* vt = (_Float16*)(ws + 6 * X_SZ);      // [48,56 MB)
    // Aliased scratch (stream-ordered; producers of the aliased regions are
    // finished with them before these are written):
    _Float16* ao   = xq;                 // attn output reuses xq region (8 MB)
    float*    proj = (float*)xk;         // fp32 proj reuses xk+xv (16 MB)

    const int XN = MROWS * D_MODEL;      // 4 Mi elements
    const int WN = D_MODEL * D_MODEL;    // 1 Mi elements

    // 0) one-shot fp32 -> f16 conversions (hoisted out of the GEMM loops)
    cvt_f16_kernel<<<XN / 2048, 256, 0, stream>>>(Q,  xq);
    cvt_f16_kernel<<<XN / 2048, 256, 0, stream>>>(K,  xk);
    cvt_f16_kernel<<<XN / 2048, 256, 0, stream>>>(V,  xv);
    cvt_f16_kernel<<<WN / 2048, 256, 0, stream>>>(Wq, wq);
    cvt_f16_kernel<<<WN / 2048, 256, 0, stream>>>(Wk, wk);
    cvt_f16_kernel<<<WN / 2048, 256, 0, stream>>>(Wv, wv);
    cvt_f16_kernel<<<WN / 2048, 256, 0, stream>>>(Wo, wo);

    // 1) Q/K/V projections (+ layout change)
    dim3 g1((MROWS / 128) * 16, 3);
    qkv_proj_kernel<<<g1, 128, 0, stream>>>(xq, xk, xv, wq, wk, wv, qh, kh, vt);

    // 2) flash attention
    attn_kernel<<<BATCH * HEADS * (SEQ / 64), 128, 0, stream>>>(qh, kh, vt, mask, ao);

    // 3) output projection
    oproj_kernel<<<(MROWS / 128) * 16, 128, 0, stream>>>(ao, wo, proj);

    // 4) residual + layernorm
    ln_kernel<<<MROWS, 256, 0, stream>>>(proj, Q, gam, bet, out);
}
